// SpikingMLP_3667902071441
// MI455X (gfx1250) — matrix operand order repaired and checked by reading, not compiled, not run on hardware
//
#include <hip/hip_runtime.h>

typedef __bf16 bf16_t;
typedef __attribute__((ext_vector_type(16))) __bf16 v16bf;
typedef __attribute__((ext_vector_type(8)))  __bf16 v8bf;
typedef __attribute__((ext_vector_type(8)))  float  v8f;

namespace {
constexpr int kT = 4;
constexpr int kM = 32 * 196;  // B*N = 6272 rows
constexpr int kC = 384;
constexpr int kH = 1536;

constexpr size_t kS1Elems = (size_t)kT * kM * kH;  // 38,535,168
constexpr size_t kXbElems = (size_t)kT * kM * kC;  //  9,633,792
constexpr size_t kW1Elems = (size_t)kH * kC;       //    589,824
constexpr size_t kW2Elems = (size_t)kC * kH;       //    589,824
}

// ---- pre-pass: fp32 -> bf16, 8 elements per thread --------------------------
__global__ __launch_bounds__(256) void cvt_f32_to_bf16(const float* __restrict__ src,
                                                       bf16_t* __restrict__ dst, int n8) {
    int i = blockIdx.x * blockDim.x + threadIdx.x;
    if (i >= n8) return;
    float4 a = ((const float4*)src)[2 * i + 0];
    float4 b = ((const float4*)src)[2 * i + 1];
    v8bf o;
    o[0] = (__bf16)a.x; o[1] = (__bf16)a.y; o[2] = (__bf16)a.z; o[3] = (__bf16)a.w;
    o[4] = (__bf16)b.x; o[5] = (__bf16)b.y; o[6] = (__bf16)b.z; o[7] = (__bf16)b.w;
    ((v8bf*)dst)[i] = o;
}

// ---- A-fragment (16x32 bf16, MxK): two contiguous 8-elem runs per lane ------
// elem j -> src[kb + j], kb = k0 + half*16
__device__ __forceinline__ v16bf load_a_bf16(const bf16_t* __restrict__ p) {
    v8bf lo = *(const v8bf*)(p);
    v8bf hi = *(const v8bf*)(p + 8);
    v16bf a;
#pragma unroll
    for (int i = 0; i < 8; ++i) { a[i] = lo[i]; a[8 + i] = hi[i]; }
    return a;
}

// ---- fused GEMM (4 time steps share the B operand) + LIF + spike store ------
// A: [T, M, K] bf16, W: [NOUT, K] bf16 (row-major over K), bias: [NOUT] f32.
// Wave tile: 16 rows x 32 cols (two 16x16 N-tiles), 4 accumulators per N-tile.
// Workgroup: 8 waves as 2(row) x 4(col) -> 32 rows x 128 cols.
template <int K, int NOUT, bool OUT_BF16>
__global__ __launch_bounds__(256) void gemm4_lif_kernel(
    const bf16_t* __restrict__ A, const bf16_t* __restrict__ W,
    const float* __restrict__ bias, void* __restrict__ outv)
{
    const int lane = threadIdx.x & 31;
    const int wave = threadIdx.x >> 5;
    const int l    = lane & 15;
    const int half = lane >> 4;
    const int row0 = blockIdx.y * 32 + (wave >> 2) * 16;
    const int col0 = blockIdx.x * 128 + (wave & 3) * 32;

    // Per-lane base pointers (K-offset folded in).
    const int kbA = half * 16;  // A: lanes 0-15 hold K 0-15; lanes 16-31 hold K 16-31
    const int kbB = half * 16;  // B: lanes 0-15 hold K 0-15; lanes 16-31 hold K 16-31
    const bf16_t* ap0 = A + ((size_t)0 * kM + row0 + l) * K + kbA;
    const bf16_t* ap1 = A + ((size_t)1 * kM + row0 + l) * K + kbA;
    const bf16_t* ap2 = A + ((size_t)2 * kM + row0 + l) * K + kbA;
    const bf16_t* ap3 = A + ((size_t)3 * kM + row0 + l) * K + kbA;
    const bf16_t* wp0 = W + (size_t)(col0 + 0  + l) * K + kbB;
    const bf16_t* wp1 = W + (size_t)(col0 + 16 + l) * K + kbB;

    v8f acc[4][2] = {};
    for (int k0 = 0; k0 < K; k0 += 32) {
        v16bf b0 = *(const v16bf*)(wp0 + k0);   // 32B contiguous
        v16bf b1 = *(const v16bf*)(wp1 + k0);
        v16bf a0 = load_a_bf16(ap0 + k0);
        acc[0][0] = __builtin_amdgcn_wmma_f32_16x16x32_bf16(false, a0, false, b0, (short)0, acc[0][0], false, false);
        acc[0][1] = __builtin_amdgcn_wmma_f32_16x16x32_bf16(false, a0, false, b1, (short)0, acc[0][1], false, false);
        v16bf a1 = load_a_bf16(ap1 + k0);
        acc[1][0] = __builtin_amdgcn_wmma_f32_16x16x32_bf16(false, a1, false, b0, (short)0, acc[1][0], false, false);
        acc[1][1] = __builtin_amdgcn_wmma_f32_16x16x32_bf16(false, a1, false, b1, (short)0, acc[1][1], false, false);
        v16bf a2 = load_a_bf16(ap2 + k0);
        acc[2][0] = __builtin_amdgcn_wmma_f32_16x16x32_bf16(false, a2, false, b0, (short)0, acc[2][0], false, false);
        acc[2][1] = __builtin_amdgcn_wmma_f32_16x16x32_bf16(false, a2, false, b1, (short)0, acc[2][1], false, false);
        v16bf a3 = load_a_bf16(ap3 + k0);
        acc[3][0] = __builtin_amdgcn_wmma_f32_16x16x32_bf16(false, a3, false, b0, (short)0, acc[3][0], false, false);
        acc[3][1] = __builtin_amdgcn_wmma_f32_16x16x32_bf16(false, a3, false, b1, (short)0, acc[3][1], false, false);
    }

    // LIF scan over t per output element: h = 0.5*(v + in); s = (h >= 1); v = h*(1-s)
    // C/D layout: col = col0 + n*16 + (lane&15), row = vgpr_i + 8*(lane>=16).
#pragma unroll
    for (int n = 0; n < 2; ++n) {
        const float bs = bias[col0 + n * 16 + l];
        float v[8];
#pragma unroll
        for (int i = 0; i < 8; ++i) v[i] = 0.0f;
#pragma unroll
        for (int t = 0; t < 4; ++t) {
            const size_t base = ((size_t)t * kM + row0 + half * 8) * NOUT + col0 + n * 16 + l;
#pragma unroll
            for (int i = 0; i < 8; ++i) {
                float in = acc[t][n][i] + bs;
                float h  = 0.5f * (v[i] + in);
                float s  = (h >= 1.0f) ? 1.0f : 0.0f;
                v[i] = h - h * s;
                if (OUT_BF16) ((bf16_t*)outv)[base + (size_t)i * NOUT] = (__bf16)s;
                else          ((float*)outv)[base + (size_t)i * NOUT]  = s;
            }
        }
    }
}

extern "C" void kernel_launch(void* const* d_in, const int* in_sizes, int n_in,
                              void* d_out, int out_size, void* d_ws, size_t ws_size,
                              hipStream_t stream) {
    const float* x  = (const float*)d_in[0];   // [4, 32, 196, 384]
    const float* W1 = (const float*)d_in[1];   // [1536, 384]
    const float* b1 = (const float*)d_in[2];   // [1536]
    const float* W2 = (const float*)d_in[3];   // [384, 1536]
    const float* b2 = (const float*)d_in[4];   // [384]
    float* out = (float*)d_out;                // [4, 32, 196, 384]

    // Workspace layout (bf16): s1 spikes | x | W1 | W2  (~99 MB total)
    bf16_t* s1  = (bf16_t*)d_ws;
    bf16_t* xb  = s1 + kS1Elems;
    bf16_t* w1b = xb + kXbElems;
    bf16_t* w2b = w1b + kW1Elems;

    // One-time fp32 -> bf16 conversion passes (amortizes cvt over all tile reuse).
    {
        int n8 = (int)(kXbElems / 8);
        cvt_f32_to_bf16<<<(n8 + 255) / 256, 256, 0, stream>>>(x, xb, n8);
        n8 = (int)(kW1Elems / 8);
        cvt_f32_to_bf16<<<(n8 + 255) / 256, 256, 0, stream>>>(W1, w1b, n8);
        n8 = (int)(kW2Elems / 8);
        cvt_f32_to_bf16<<<(n8 + 255) / 256, 256, 0, stream>>>(W2, w2b, n8);
    }

    dim3 blk(256);
    // fc1 + LIF: [M x C] @ [C x H] -> spikes s1 (bf16, exact)
    gemm4_lif_kernel<kC, kH, true><<<dim3(kH / 128, kM / 32), blk, 0, stream>>>(xb, w1b, b1, s1);
    // fc2 + LIF: [M x H] @ [H x C] -> spikes out (fp32)
    gemm4_lif_kernel<kH, kC, false><<<dim3(kC / 128, kM / 32), blk, 0, stream>>>(s1, w2b, b2, out);
}
